// BRCLoss_89678917140919
// MI455X (gfx1250) — matrix operand order repaired and checked by reading, not compiled, not run on hardware
//
#include <hip/hip_runtime.h>
#include <hip/hip_bf16.h>
#include <math.h>

// ---- types for WMMA fragments (CDNA5 wave32) ----
typedef __bf16 bf16x4 __attribute__((ext_vector_type(4)));
typedef __bf16 bf16x8 __attribute__((ext_vector_type(8)));
typedef __bf16 v16bf  __attribute__((ext_vector_type(16)));
typedef float  v8f    __attribute__((ext_vector_type(8)));

#define NROW   24        // N_VIEWS * NUM_CLASS anchors per sample
#define MPAD   32        // padded to 2x2 WMMA tiles
#define DDIM   4096
#define CH     256       // K-chunk staged in LDS
#define STRIDE 264       // LDS row stride (bf16) -> 528B, rotates banks, 16B aligned
#define NCHUNK (DDIM / CH)          // 16
#define F4PERCHUNK (NROW * (CH/4))  // 1536 float4 per chunk
#define LDIT   (F4PERCHUNK / 256)   // 6 staged loads per thread per chunk

__global__ __launch_bounds__(256)
void brc_loss_kernel(const float* __restrict__ features, float* __restrict__ out) {
  __shared__ __bf16 sbuf[2][MPAD * STRIDE];   // double-buffered bf16 feature tile
  __shared__ float  G[MPAD][MPAD + 1];        // raw Gram (padded)
  __shared__ float  invn[MPAD];
  __shared__ float  rowloss[NROW];

  const int tid   = threadIdx.x;
  const int lane  = tid & 31;
  const int w     = tid >> 5;        // wave 0..7
  const int tile  = w & 3;           // which 16x16 tile of 32x32 Gram
  const int tm    = tile >> 1;
  const int tn    = tile & 1;
  const int khalf = w >> 2;          // K-split across wave pairs
  const int lh    = lane >> 4;       // lane half (0/1)
  const int lm    = lane & 15;

  // Zero padded rows 24..31 in both buffers (loads never touch them).
  #pragma unroll
  for (int it = 0; it < (2 * 8 * (CH / 4)) / 256; ++it) {  // 1024 bf16x4 stores
    int e    = tid + it * 256;
    int bufi = e >> 9;              // / 512
    int rem  = e & 511;
    int row  = NROW + (rem >> 6);   // 64 bf16x4 per row
    int c4   = rem & 63;
    bf16x4 z = {(__bf16)0.f, (__bf16)0.f, (__bf16)0.f, (__bf16)0.f};
    *(bf16x4*)(&sbuf[bufi][row * STRIDE + (c4 << 2)]) = z;
  }

  const float4* F4 = reinterpret_cast<const float4*>(
      features + (size_t)blockIdx.x * NROW * DDIM);

  v8f acc = {0.f, 0.f, 0.f, 0.f, 0.f, 0.f, 0.f, 0.f};

  const int arow = (tm * 16 + lm) * STRIDE;  // A-frag row (M)
  const int brow = (tn * 16 + lm) * STRIDE;  // B-frag row = column of G (F^T)

  union Frag { bf16x8 h[2]; v16bf v; };

  for (int c = 0; c < NCHUNK; ++c) {
    // ---- stage chunk c: batch all global loads, then convert fp32->bf16 to LDS
    __bf16* dst = sbuf[c & 1];
    float4 v[LDIT];
    #pragma unroll
    for (int it = 0; it < LDIT; ++it) {
      int e   = tid + it * 256;
      int row = e >> 6;             // CH/4 = 64 float4 per row
      int c4  = e & 63;
      v[it] = F4[row * (DDIM / 4) + c * (CH / 4) + c4];
    }
    #pragma unroll
    for (int it = 0; it < LDIT; ++it) {
      int e   = tid + it * 256;
      int row = e >> 6;
      int c4  = e & 63;
      bf16x4 o = {(__bf16)v[it].x, (__bf16)v[it].y, (__bf16)v[it].z, (__bf16)v[it].w};
      *(bf16x4*)(dst + row * STRIDE + (c4 << 2)) = o;
    }
    __syncthreads();   // single barrier per chunk (double buffer covers WAR)

    // ---- fragment loads for all 4 K-steps issued up front (16x ds_load_b128),
    //      then 4 chained WMMAs; staged dscnt waits overlap LDS latency.
    const __bf16* base = sbuf[c & 1];
    Frag A[4], B[4];
    #pragma unroll
    for (int s = 0; s < 4; ++s) {
      const int k0 = (khalf * 4 + s) * 32;
      // A 16x32 bf16 layout: lanes 0-15 hold K {0..7,16..23}, lanes 16-31 {8..15,24..31}
      A[s].h[0] = *(const bf16x8*)(base + arow + k0 + lh * 8);
      A[s].h[1] = *(const bf16x8*)(base + arow + k0 + lh * 8 + 16);
      // B 32x16 bf16 layout: lanes 0-15 hold K {0..7,16..23}, lanes 16-31 {8..15,24..31}
      B[s].h[0] = *(const bf16x8*)(base + brow + k0 + lh * 8);
      B[s].h[1] = *(const bf16x8*)(base + brow + k0 + lh * 8 + 16);
    }
    #pragma unroll
    for (int s = 0; s < 4; ++s) {
      acc = __builtin_amdgcn_wmma_f32_16x16x32_bf16(
          false, A[s].v, false, B[s].v, (short)0, acc, false, false);
    }
  }
  __syncthreads();

  // ---- merge K-halves and materialize Gram in LDS (C layout: lane half -> M+8)
  const int m0 = tm * 16 + lh * 8;
  const int n  = tn * 16 + lm;
  if (w < 4) {
    #pragma unroll
    for (int r = 0; r < 8; ++r) G[m0 + r][n] = acc[r];
  }
  __syncthreads();
  if (w >= 4) {
    #pragma unroll
    for (int r = 0; r < 8; ++r) G[m0 + r][n] += acc[r];
  }
  __syncthreads();

  // ---- inverse norms from Gram diagonal (F.normalize eps = 1e-12)
  if (tid < MPAD) {
    float s = sqrtf(G[tid][tid]);
    invn[tid] = 1.0f / fmaxf(s, 1e-12f);
  }
  __syncthreads();

  // ---- per-anchor masked log-softmax (unstable exp, matching reference)
  if (tid < NROW) {
    const int   i     = tid;
    const int   ci    = (i >= 12) ? i - 12 : i;
    const float inv_i = invn[i];
    float sumexp = 0.f, possum = 0.f;
    int   npos   = 0;
    for (int j = 0; j < NROW; ++j) {
      if (j == i) continue;
      const int   cj = (j >= 12) ? j - 12 : j;
      const float lg = G[i][j] * inv_i * invn[j] * 10.0f;  // / TEMPERATURE
      sumexp += __expf(lg);
      int dc = ci - cj; if (dc < 0) dc = -dc;
      if (dc <= 1) { possum += lg; ++npos; }
    }
    const float logden = __logf(sumexp);
    rowloss[i] = (possum - (float)npos * logden) / ((float)npos + 1e-6f);
  }
  __syncthreads();

  if (tid == 0) {
    float s = 0.f;
    #pragma unroll
    for (int i = 0; i < NROW; ++i) s += rowloss[i];
    // -(T/baseT) * mean over 24 anchors, summed over B then / B
    atomicAdd(out, s * (-0.1f / (24.0f * 512.0f)));
  }
}

extern "C" void kernel_launch(void* const* d_in, const int* in_sizes, int n_in,
                              void* d_out, int out_size, void* d_ws, size_t ws_size,
                              hipStream_t stream) {
  (void)n_in; (void)d_ws; (void)ws_size;
  // inputs: [0] outputs (unused), [1] targets (unused), [2] features
  const float* features = (const float*)d_in[2];
  float* out = (float*)d_out;
  const int B = in_sizes[2] / (NROW * DDIM);   // 512
  hipMemsetAsync(out, 0, (size_t)out_size * sizeof(float), stream);
  brc_loss_kernel<<<B, 256, 0, stream>>>(features, out);
}
